// Attention_28630251995211
// MI455X (gfx1250) — compile-verified, matrix operand order checked by reading, not run on hardware
//
#include <hip/hip_runtime.h>
#include <math.h>

typedef _Float16 h8  __attribute__((ext_vector_type(8)));
typedef _Float16 h16 __attribute__((ext_vector_type(16)));
typedef float    f8  __attribute__((ext_vector_type(8)));

#define BB   2
#define SS   2048
#define DIMD 2048
#define NH   16
#define NKV  4
#define HDIM 128
#define GQ   4

#define WMMA_F16(a, b, c) \
  __builtin_amdgcn_wmma_f32_16x16x32_f16(false, (a), false, (b), (short)0, (c), false, false)

union H16U { h16 v; h8 h[2]; };

// Load a 16x32 f16 A-fragment (or a 32x16 B-fragment, which mirrors it):
// lane l -> row (l&15); K elements j<8 at K = j + 8*(l>>4), j>=8 at K = j + 8 + 8*(l>>4).
// Per lane: two contiguous 16-byte loads.
static __device__ __forceinline__ h16 load_frag16(const _Float16* __restrict__ base,
                                                  int strideHalves) {
  int lane = threadIdx.x & 31;
  const _Float16* p = base + (size_t)(lane & 15) * strideHalves + 8 * (lane >> 4);
  H16U u;
  u.h[0] = *(const h8*)(p);
  u.h[1] = *(const h8*)(p + 16);
  return u.v;
}

// ---------------------------------------------------------------- conversions
__global__ void f32_to_f16_kernel(const float* __restrict__ src,
                                  _Float16* __restrict__ dst, size_t n) {
  size_t i = (size_t)blockIdx.x * blockDim.x + threadIdx.x;
  if (i < n) dst[i] = (_Float16)src[i];
}

// ---------------------------------------------------------------- GEMM
// C[M,N] (f32) = A[M,K] (f16, row-major) * B[N,K]^T (f16, row-major rows of K)
// 128x128 workgroup tile, K-chunks of 32, double-buffered LDS, software pipelined.
#define MT 128
#define NT 128
#define KT 32
#define LSTR 48   // LDS row stride in halves (96B, 16B-aligned, conflict padding)

__global__ __launch_bounds__(256)
void gemm_f16_wmma(const _Float16* __restrict__ A, const _Float16* __restrict__ Bw,
                   float* __restrict__ C, int M, int N, int K) {
  __shared__ __align__(16) _Float16 Al[2][MT * LSTR];
  __shared__ __align__(16) _Float16 Bl[2][NT * LSTR];
  int tid  = threadIdx.x;
  int wave = tid >> 5;
  int lane = tid & 31;
  int m0 = blockIdx.y * MT;
  int n0 = blockIdx.x * NT;
  int wm = 64 * (wave >> 2);   // wave M offset: 0 or 64   (4 x 16-row tiles)
  int wn = 32 * (wave & 3);    // wave N offset: 0..96     (2 x 16-col tiles)

  f8 acc[4][2];
  for (int am = 0; am < 4; ++am)
    for (int bn = 0; bn < 2; ++bn) acc[am][bn] = {};

  int sr = tid >> 1;           // staged row 0..127
  int sc = (tid & 1) * 16;     // staged col: halves 0..15 or 16..31
  const _Float16* ga = A  + (size_t)(m0 + sr) * K + sc;
  const _Float16* gb = Bw + (size_t)(n0 + sr) * K + sc;

  // prologue: stage chunk 0 into buffer 0
  h8 ra0 = *(const h8*)(ga);
  h8 ra1 = *(const h8*)(ga + 8);
  h8 rb0 = *(const h8*)(gb);
  h8 rb1 = *(const h8*)(gb + 8);
  *(h8*)&Al[0][sr * LSTR + sc]     = ra0;
  *(h8*)&Al[0][sr * LSTR + sc + 8] = ra1;
  *(h8*)&Bl[0][sr * LSTR + sc]     = rb0;
  *(h8*)&Bl[0][sr * LSTR + sc + 8] = rb1;
  __syncthreads();

  int nk = K / KT;
  for (int kk = 0; kk < nk; ++kk) {
    int cur = kk & 1;
    // issue next chunk's global loads before computing (latency hidden by WMMAs)
    if (kk + 1 < nk) {
      const _Float16* pa = ga + (size_t)(kk + 1) * KT;
      const _Float16* pb = gb + (size_t)(kk + 1) * KT;
      ra0 = *(const h8*)(pa);
      ra1 = *(const h8*)(pa + 8);
      rb0 = *(const h8*)(pb);
      rb1 = *(const h8*)(pb + 8);
    }
    if (kk + 2 < nk) {  // warm L2/L0 two chunks ahead (global_prefetch_b8)
      __builtin_prefetch((const void*)(ga + (size_t)(kk + 2) * KT), 0, 3);
      __builtin_prefetch((const void*)(gb + (size_t)(kk + 2) * KT), 0, 3);
    }
    // compute on current buffer: 8 WMMAs per wave per chunk
    h16 af[4];
    for (int am = 0; am < 4; ++am)
      af[am] = load_frag16(&Al[cur][(wm + 16 * am) * LSTR], LSTR);
    for (int bn = 0; bn < 2; ++bn) {
      h16 bf = load_frag16(&Bl[cur][(wn + 16 * bn) * LSTR], LSTR);
      for (int am = 0; am < 4; ++am)
        acc[am][bn] = WMMA_F16(af[am], bf, acc[am][bn]);
    }
    // drain staged regs into the other buffer (readers of it synced last iter)
    if (kk + 1 < nk) {
      int nxt = cur ^ 1;
      *(h8*)&Al[nxt][sr * LSTR + sc]     = ra0;
      *(h8*)&Al[nxt][sr * LSTR + sc + 8] = ra1;
      *(h8*)&Bl[nxt][sr * LSTR + sc]     = rb0;
      *(h8*)&Bl[nxt][sr * LSTR + sc + 8] = rb1;
    }
    __syncthreads();
  }

  int col = lane & 15, hf = lane >> 4;
  for (int am = 0; am < 4; ++am)
    for (int bn = 0; bn < 2; ++bn)
      for (int r = 0; r < 8; ++r) {
        int row = m0 + wm + 16 * am + r + 8 * hf;
        C[(size_t)row * N + n0 + wn + 16 * bn + col] = acc[am][bn][r];
      }
}

// ---------------------------------------------------------------- RoPE + repack
// proj: [BB*SS, nh*HDIM] f32 -> out: [BB, nh, SS, HDIM] f16 (optionally pre-scaled)
__global__ void rope_pack_kernel(const float* __restrict__ proj,
                                 const float* __restrict__ freqs,
                                 _Float16* __restrict__ out, int nh, float scale) {
  size_t idx = (size_t)blockIdx.x * blockDim.x + threadIdx.x;
  size_t total = (size_t)BB * SS * nh * (HDIM / 2);
  if (idx >= total) return;
  int d2 = (int)(idx % (HDIM / 2));
  size_t t = idx / (HDIM / 2);
  int h = (int)(t % nh); t /= nh;
  int s = (int)(t % SS);
  int b = (int)(t / SS);
  const float* pr = proj + (size_t)(b * SS + s) * ((size_t)nh * HDIM) + (size_t)h * HDIM + 2 * d2;
  float x0 = pr[0], x1 = pr[1];
  float c  = freqs[(size_t)s * HDIM + 2 * d2];
  float sn = freqs[(size_t)s * HDIM + 2 * d2 + 1];
  float o0 = (x0 * c - x1 * sn) * scale;
  float o1 = (x0 * sn + x1 * c) * scale;
  _Float16* po = out + ((size_t)(b * nh + h) * SS + s) * HDIM + 2 * d2;
  po[0] = (_Float16)o0;
  po[1] = (_Float16)o1;
}

// V: [BB*SS, NKV*HDIM] f32 -> Vt: [BB, NKV, HDIM, SS] f16
__global__ void pack_vt_kernel(const float* __restrict__ v, _Float16* __restrict__ vt) {
  size_t idx = (size_t)blockIdx.x * blockDim.x + threadIdx.x;
  size_t total = (size_t)BB * SS * NKV * HDIM;
  if (idx >= total) return;
  int d = (int)(idx % HDIM);
  size_t t = idx / HDIM;
  int kv = (int)(t % NKV); t /= NKV;
  int s = (int)(t % SS);
  int b = (int)(t / SS);
  float val = v[(size_t)(b * SS + s) * (NKV * HDIM) + kv * HDIM + d];
  vt[((size_t)(b * NKV + kv) * HDIM + d) * SS + s] = (_Float16)val;
}

// ---------------------------------------------------------------- flash attention
// Q: [BB,NH,SS,HDIM] f16 (pre-scaled by 1/sqrt(HDIM)); K: [BB,NKV,SS,HDIM] f16;
// Vt: [BB,NKV,HDIM,SS] f16;  O: [BB,SS,NH*HDIM] f16.  One wave per 16-row Q tile.
__global__ __launch_bounds__(32)
void flash_attn_kernel(const _Float16* __restrict__ Q, const _Float16* __restrict__ Kc,
                       const _Float16* __restrict__ Vt, _Float16* __restrict__ O) {
  __shared__ __align__(16) _Float16 Pl[16 * 32];
  int lane = threadIdx.x & 31;
  int col = lane & 15, hf = lane >> 4;
  int q0 = blockIdx.x * 16;
  int h  = blockIdx.y;
  int b  = blockIdx.z;
  int kv = h / GQ;
  const _Float16* qbase = Q  + ((size_t)(b * NH + h) * SS + q0) * HDIM;
  const _Float16* kbase = Kc + ((size_t)(b * NKV + kv) * SS) * HDIM;
  const _Float16* vbase = Vt + ((size_t)(b * NKV + kv) * HDIM) * SS;

  h16 qf[4];
  for (int kc = 0; kc < 4; ++kc) qf[kc] = load_frag16(qbase + 32 * kc, HDIM);

  f8 oacc[8];
  for (int nd = 0; nd < 8; ++nd) oacc[nd] = {};
  float mrow[8], lrow[8];
  for (int r = 0; r < 8; ++r) { mrow[r] = -__builtin_inff(); lrow[r] = 0.0f; }

  int jend = (q0 + 15) / 32;
  for (int j = 0; j <= jend; ++j) {
    int kb = 32 * j;
    if (j < jend) {  // prefetch next K block (global_prefetch_b8)
      __builtin_prefetch((const void*)(kbase + (size_t)(kb + 32) * HDIM), 0, 3);
    }
    // S tile 16x32 = two 16x16 WMMAs over 4 K-chunks of the head dim
    f8 s0 = {}, s1 = {};
    for (int kc = 0; kc < 4; ++kc) {
      h16 b0 = load_frag16(kbase + (size_t)kb * HDIM + 32 * kc, HDIM);
      h16 b1 = load_frag16(kbase + (size_t)(kb + 16) * HDIM + 32 * kc, HDIM);
      s0 = WMMA_F16(qf[kc], b0, s0);
      s1 = WMMA_F16(qf[kc], b1, s1);
    }
    // online softmax per row (rows live across lanes 0-15 / 16-31 per half)
    for (int r = 0; r < 8; ++r) {
      int qi = q0 + r + 8 * hf;
      float v0 = (kb + col)      <= qi ? s0[r] : -__builtin_inff();
      float v1 = (kb + 16 + col) <= qi ? s1[r] : -__builtin_inff();
      float mx = fmaxf(v0, v1);
      for (int msk = 1; msk < 16; msk <<= 1) mx = fmaxf(mx, __shfl_xor(mx, msk, 32));
      float mnew  = fmaxf(mrow[r], mx);
      float alpha = expf(mrow[r] - mnew);
      float p0 = expf(v0 - mnew);
      float p1 = expf(v1 - mnew);
      float ps = p0 + p1;
      for (int msk = 1; msk < 16; msk <<= 1) ps += __shfl_xor(ps, msk, 32);
      lrow[r] = lrow[r] * alpha + ps;
      mrow[r] = mnew;
      for (int nd = 0; nd < 8; ++nd) oacc[nd][r] *= alpha;
      Pl[(r + 8 * hf) * 32 + col]      = (_Float16)p0;
      Pl[(r + 8 * hf) * 32 + 16 + col] = (_Float16)p1;
    }
    __syncthreads();                 // C-layout -> A-layout via LDS
    h16 pf = load_frag16(Pl, 32);
    for (int nd = 0; nd < 8; ++nd) {
      h16 vb = load_frag16(vbase + (size_t)(16 * nd) * SS + kb, SS);
      oacc[nd] = WMMA_F16(pf, vb, oacc[nd]);
    }
    __syncthreads();
  }
  for (int nd = 0; nd < 8; ++nd)
    for (int r = 0; r < 8; ++r) {
      int qi = q0 + r + 8 * hf;
      float val = oacc[nd][r] / lrow[r];
      O[((size_t)(b * SS + qi) * NH + h) * HDIM + 16 * nd + col] = (_Float16)val;
    }
}

// ---------------------------------------------------------------- launch
extern "C" void kernel_launch(void* const* d_in, const int* in_sizes, int n_in,
                              void* d_out, int out_size, void* d_ws, size_t ws_size,
                              hipStream_t stream) {
  (void)in_sizes; (void)n_in; (void)out_size; (void)ws_size;
  const float* x     = (const float*)d_in[0];
  const float* freqs = (const float*)d_in[1];
  const float* wq    = (const float*)d_in[2];
  const float* wk    = (const float*)d_in[3];
  const float* wv    = (const float*)d_in[4];
  const float* wo    = (const float*)d_in[5];
  float* out = (float*)d_out;

  const size_t M = (size_t)BB * SS;  // 4096 tokens
  char* ws = (char*)d_ws;
  size_t off = 0;
  auto take = [&](size_t bytes) -> void* {
    void* p = ws + off;
    off = (off + bytes + 255) & ~(size_t)255;
    return p;
  };
  _Float16* x_h  = (_Float16*)take(M * DIMD * 2);
  _Float16* wq_h = (_Float16*)take((size_t)NH * HDIM * DIMD * 2);
  _Float16* wk_h = (_Float16*)take((size_t)NKV * HDIM * DIMD * 2);
  _Float16* wv_h = (_Float16*)take((size_t)NKV * HDIM * DIMD * 2);
  _Float16* wo_h = (_Float16*)take((size_t)DIMD * NH * HDIM * 2);
  float* q_f = (float*)take(M * (size_t)NH * HDIM * 4);
  float* k_f = (float*)take(M * (size_t)NKV * HDIM * 4);
  float* v_f = (float*)take(M * (size_t)NKV * HDIM * 4);
  _Float16* q_h  = (_Float16*)take((size_t)BB * NH * SS * HDIM * 2);
  _Float16* k_h  = (_Float16*)take((size_t)BB * NKV * SS * HDIM * 2);
  _Float16* vt_h = (_Float16*)take((size_t)BB * NKV * HDIM * SS * 2);
  _Float16* attn_h = (_Float16*)q_f;  // reuse dead q_f32 region (16MB < 32MB)

  auto cvt = [&](const float* s, _Float16* d, size_t n) {
    f32_to_f16_kernel<<<dim3((unsigned)((n + 255) / 256)), dim3(256), 0, stream>>>(s, d, n);
  };
  cvt(x,  x_h,  M * DIMD);
  cvt(wq, wq_h, (size_t)NH * HDIM * DIMD);
  cvt(wk, wk_h, (size_t)NKV * HDIM * DIMD);
  cvt(wv, wv_h, (size_t)NKV * HDIM * DIMD);
  cvt(wo, wo_h, (size_t)DIMD * NH * HDIM);

  // Projections: C = x @ W^T
  gemm_f16_wmma<<<dim3(NH * HDIM / NT, (unsigned)(M / MT)), dim3(256), 0, stream>>>(
      x_h, wq_h, q_f, (int)M, NH * HDIM, DIMD);
  gemm_f16_wmma<<<dim3(NKV * HDIM / NT, (unsigned)(M / MT)), dim3(256), 0, stream>>>(
      x_h, wk_h, k_f, (int)M, NKV * HDIM, DIMD);
  gemm_f16_wmma<<<dim3(NKV * HDIM / NT, (unsigned)(M / MT)), dim3(256), 0, stream>>>(
      x_h, wv_h, v_f, (int)M, NKV * HDIM, DIMD);

  // RoPE + layout pack (softmax scale folded into Q)
  {
    size_t nq = (size_t)BB * SS * NH * (HDIM / 2);
    rope_pack_kernel<<<dim3((unsigned)((nq + 255) / 256)), dim3(256), 0, stream>>>(
        q_f, freqs, q_h, NH, 0.08838834764831845f /* 1/sqrt(128) */);
    size_t nk = (size_t)BB * SS * NKV * (HDIM / 2);
    rope_pack_kernel<<<dim3((unsigned)((nk + 255) / 256)), dim3(256), 0, stream>>>(
        k_f, freqs, k_h, NKV, 1.0f);
    size_t nv = (size_t)BB * SS * NKV * HDIM;
    pack_vt_kernel<<<dim3((unsigned)((nv + 255) / 256)), dim3(256), 0, stream>>>(v_f, vt_h);
  }

  // Causal GQA flash attention
  flash_attn_kernel<<<dim3(SS / 16, NH, BB), dim3(32), 0, stream>>>(q_h, k_h, vt_h, attn_h);

  // Output projection: out = attn @ wo^T  (f32 straight into d_out)
  gemm_f16_wmma<<<dim3(DIMD / NT, (unsigned)(M / MT)), dim3(256), 0, stream>>>(
      attn_h, wo_h, out, (int)M, DIMD, NH * HDIM);
}
